// GRU_67611375174162
// MI455X (gfx1250) — matrix operand order repaired and checked by reading, not compiled, not run on hardware
//
#include <hip/hip_runtime.h>
#include <hip/hip_bf16.h>
#include <stdint.h>

// ---------------------------------------------------------------------------
// GRU on MI455X (gfx1250): bf16 WMMA + fp32 accumulate.
//  - Wave computes a 64x16 output column: 4 independent WMMA accumulator
//    chains share one B fragment per K-step (4x B-traffic cut + WMMA ILP).
//  - Whole A panel (64xK bf16) staged global->LDS ONCE per block with async
//    copies (global_load_async_to_lds_b128, one s_wait_asynccnt, one barrier),
//    padded to an 80B row stride for conflict-free ds_load_b128.
//  - Barrier-free K-loop afterwards: scheduler pipelines B loads + LDS reads
//    against WMMA without accumulator shuffling.
//  - Weights bf16 (9 MB) stay L2-resident across the 512-step scan.
// ---------------------------------------------------------------------------

static constexpr int Tn = 512;
static constexpr int Bn = 64;
static constexpr int In = 512;
static constexpr int Hn = 1024;

static constexpr int ROWS = 40;          // halves per staged row (32 + 8 pad)
static constexpr int CHUNK = 64 * ROWS;  // halves per staged 64x32 chunk

typedef __attribute__((ext_vector_type(16))) __bf16   v16bf;
typedef __attribute__((ext_vector_type(8)))  float    v8f;
typedef __attribute__((ext_vector_type(4)))  uint32_t u32x4;

union FragU { v16bf v; u32x4 u[2]; };

__device__ __forceinline__ uint16_t f32_to_bf16_bits(float f) {
  union { float f; uint32_t u; } c; c.f = f;
  uint32_t u = c.u;
  u += 0x7fffu + ((u >> 16) & 1u);      // round-to-nearest-even
  return (uint16_t)(u >> 16);
}

// Async 16B global -> LDS copy (per lane). ASYNCcnt-tracked.
__device__ __forceinline__ void async_stage16(const uint16_t* src_g,
                                              uint16_t* dst_lds) {
  uint32_t lds_off = (uint32_t)(uintptr_t)dst_lds;   // AS(3) offset in low bits
  asm volatile("global_load_async_to_lds_b128 %0, %1, off"
               :: "v"(lds_off), "v"(src_g) : "memory");
}
__device__ __forceinline__ void wait_async0() {
  asm volatile("s_wait_asynccnt 0x0" ::: "memory");
}

// B fragment (32x16 bf16). out = a @ W.T => B column n == W row n: each lane
// reads 8 bf16 from W[n][k0 + 8g ...] and 8 from W[n][k0 + 16 + 8g ...].
__device__ __forceinline__ v16bf load_frag_b(const uint16_t* __restrict__ W,
                                             int ldb, int n0, int k0, int lane) {
  const int n = n0 + (lane & 15);
  const int g = lane >> 4;
  const uint16_t* p = W + (size_t)n * ldb + k0 + 8 * g;
  FragU r;
  r.u[0] = *(const u32x4*)(p);
  r.u[1] = *(const u32x4*)(p + 16);
  __builtin_prefetch((const void*)(p + 32), 0, 3);   // global_prefetch_b8
  return r.v;
}

// A fragment from a staged 64x32 chunk (row stride ROWS halves = 80B).
__device__ __forceinline__ v16bf frag_a_lds(const uint16_t* cbase, int mt, int lane) {
  const int m = mt * 16 + (lane & 15);
  const int g = lane >> 4;
  const uint16_t* p = cbase + m * ROWS + 8 * g;
  FragU r;
  r.u[0] = *(const u32x4*)(p);
  r.u[1] = *(const u32x4*)(p + 16);
  return r.v;
}

__device__ __forceinline__ v8f wmma_bf16(v16bf a, v16bf b, v8f c) {
  return __builtin_amdgcn_wmma_f32_16x16x32_bf16(false, a, false, b,
                                                 (short)0, c, false, false);
}

// 64xK (A, global, stride lda) @ Kx16 (cols n0..n0+15 of W.T) -> acc[4].
// Entire A panel staged to LDS up front; barrier-free WMMA main loop.
template <int NCHUNK>
__device__ __forceinline__ void gemm_col64(
    const uint16_t* __restrict__ A, int lda,
    const uint16_t* __restrict__ W, int ldb, int n0,
    uint16_t* panel, v8f acc[4], int tid, int lane) {
  const int sm = tid >> 2;                  // staged row 0..63
  const int sq = tid & 3;                   // 16B quarter within row
  const uint16_t* src = A + (size_t)sm * lda + sq * 8;
  uint16_t*       dst = panel + sm * ROWS + sq * 8;
#pragma unroll
  for (int i = 0; i < NCHUNK; ++i)
    async_stage16(src + i * 32, dst + i * CHUNK);
  wait_async0();
  __syncthreads();
#pragma unroll
  for (int i = 0; i < NCHUNK; ++i) {
    v16bf b = load_frag_b(W, ldb, n0, i * 32, lane);
    const uint16_t* cbase = panel + i * CHUNK;
#pragma unroll
    for (int j = 0; j < 4; ++j) {           // 4 independent WMMA chains
      v16bf a = frag_a_lds(cbase, j, lane);
      acc[j] = wmma_bf16(a, b, acc[j]);
    }
  }
}

// ---------------------------------------------------------------------------
__global__ void k_f32_to_bf16(const float* __restrict__ in,
                              uint16_t* __restrict__ out, int n) {
  int i = blockIdx.x * blockDim.x + threadIdx.x;
  int stride = gridDim.x * blockDim.x;
  for (; i < n; i += stride) out[i] = f32_to_bf16_bits(in[i]);
}

// Phase 1: xg = x @ W_x2g.T + b_x2g, g in {z,r,h}. M=T*B, K=I, N=H.
// Block = 8 waves = 8 N-tiles of a shared 64-row A panel (staged once).
__global__ __launch_bounds__(256) void k_xproj(
    const uint16_t* __restrict__ Xb,
    const uint16_t* __restrict__ Wz, const uint16_t* __restrict__ Wr,
    const uint16_t* __restrict__ Wh,
    const float* __restrict__ bz, const float* __restrict__ br,
    const float* __restrict__ bh,
    float* __restrict__ oz, float* __restrict__ orr, float* __restrict__ oh) {
  __shared__ __align__(16) uint16_t panel[(In / 32) * CHUNK];   // 80 KB
  const int tid  = threadIdx.x;
  const int lane = tid & 31;
  const int w    = tid >> 5;
  const int gate = blockIdx.z;
  const int m_base = blockIdx.x * 64;
  const int n0     = (blockIdx.y * 8 + w) * 16;

  const uint16_t* Wg = (gate == 0) ? Wz : ((gate == 1) ? Wr : Wh);
  const float*    bg = (gate == 0) ? bz : ((gate == 1) ? br : bh);
  float*          og = (gate == 0) ? oz : ((gate == 1) ? orr : oh);

  v8f acc[4] = {};
  gemm_col64<In / 32>(Xb + (size_t)m_base * In, In, Wg, In, n0,
                      panel, acc, tid, lane);

  const int g = lane >> 4;
  const int n = n0 + (lane & 15);
  const float bias = bg[n];
#pragma unroll
  for (int j = 0; j < 4; ++j)
    for (int v = 0; v < 8; ++v) {
      const int m = m_base + j * 16 + v + 8 * g;
      og[(size_t)m * Hn + n] = acc[j][v] + bias;
    }
}

// Phase 2a: z = sig(xz_t + h@Wz.T + bz); rh = sig(xr_t + h@Wr.T + br) * h.
// grid(8,2): blockIdx.y = gate. Wave covers all 64 batch rows of one N-tile.
__global__ __launch_bounds__(256) void k_step_zr(
    const uint16_t* __restrict__ h_bf, const float* __restrict__ h,
    const uint16_t* __restrict__ Wz, const uint16_t* __restrict__ Wr,
    const float* __restrict__ bz, const float* __restrict__ br,
    const float* __restrict__ xz_t, const float* __restrict__ xr_t,
    float* __restrict__ z_out, uint16_t* __restrict__ rh_out) {
  __shared__ __align__(16) uint16_t panel[(Hn / 32) * CHUNK];   // 160 KB
  const int tid  = threadIdx.x;
  const int lane = tid & 31;
  const int w    = tid >> 5;
  const int n0   = (blockIdx.x * 8 + w) * 16;
  const int gate = blockIdx.y;

  const uint16_t* Wg = gate ? Wr : Wz;
  const float*    bg = gate ? br : bz;
  const float*    xg = gate ? xr_t : xz_t;

  v8f acc[4] = {};
  gemm_col64<Hn / 32>(h_bf, Hn, Wg, Hn, n0, panel, acc, tid, lane);

  const int g = lane >> 4;
  const int n = n0 + (lane & 15);
  const float bias = bg[n];
#pragma unroll
  for (int j = 0; j < 4; ++j)
    for (int v = 0; v < 8; ++v) {
      const int m = j * 16 + v + 8 * g;
      const size_t idx = (size_t)m * Hn + n;
      const float pre = acc[j][v] + bias + xg[idx];
      const float s = 1.0f / (1.0f + __expf(-pre));
      if (gate == 0) z_out[idx] = s;
      else           rh_out[idx] = f32_to_bf16_bits(s * h[idx]);
    }
}

// Phase 2b: hc = tanh(xh_t + rh@Wh.T + bh); h = (1-z)*h + z*hc; emit out[t].
__global__ __launch_bounds__(256) void k_step_h(
    const uint16_t* __restrict__ rh_bf, const uint16_t* __restrict__ Wh,
    const float* __restrict__ bh, const float* __restrict__ xh_t,
    const float* __restrict__ z, float* __restrict__ h,
    uint16_t* __restrict__ h_bf, float* __restrict__ out_t) {
  __shared__ __align__(16) uint16_t panel[(Hn / 32) * CHUNK];   // 160 KB
  const int tid  = threadIdx.x;
  const int lane = tid & 31;
  const int w    = tid >> 5;
  const int n0   = (blockIdx.x * 8 + w) * 16;

  v8f acc[4] = {};
  gemm_col64<Hn / 32>(rh_bf, Hn, Wh, Hn, n0, panel, acc, tid, lane);

  const int g = lane >> 4;
  const int n = n0 + (lane & 15);
  const float bias = bh[n];
#pragma unroll
  for (int j = 0; j < 4; ++j)
    for (int v = 0; v < 8; ++v) {
      const int m = j * 16 + v + 8 * g;
      const size_t idx = (size_t)m * Hn + n;
      const float hc = tanhf(acc[j][v] + bias + xh_t[idx]);
      const float zz = z[idx];
      const float hn = (1.0f - zz) * h[idx] + zz * hc;
      h[idx]     = hn;
      h_bf[idx]  = f32_to_bf16_bits(hn);
      out_t[idx] = hn;
    }
}

// ---------------------------------------------------------------------------
extern "C" void kernel_launch(void* const* d_in, const int* in_sizes, int n_in,
                              void* d_out, int out_size, void* d_ws, size_t ws_size,
                              hipStream_t stream) {
  const float* x   = (const float*)d_in[0];
  const float* h0  = (const float*)d_in[1];
  const float* Wxz = (const float*)d_in[2];
  const float* bxz = (const float*)d_in[3];
  const float* Whz = (const float*)d_in[4];
  const float* bhz = (const float*)d_in[5];
  const float* Wxr = (const float*)d_in[6];
  const float* bxr = (const float*)d_in[7];
  const float* Whr = (const float*)d_in[8];
  const float* bhr = (const float*)d_in[9];
  const float* Wxh = (const float*)d_in[10];
  const float* bxh = (const float*)d_in[11];
  const float* Whh = (const float*)d_in[12];
  const float* bhh = (const float*)d_in[13];
  float* out = (float*)d_out;
  (void)in_sizes; (void)n_in; (void)out_size; (void)ws_size;

  char* ws = (char*)d_ws;
  size_t off = 0;
  auto carve = [&](size_t bytes) -> char* {
    char* p = ws + off;
    off = (off + bytes + 255) & ~(size_t)255;
    return p;
  };
  uint16_t* x_bf   = (uint16_t*)carve((size_t)Tn * Bn * In * 2);
  uint16_t* Wxz_bf = (uint16_t*)carve((size_t)Hn * In * 2);
  uint16_t* Wxr_bf = (uint16_t*)carve((size_t)Hn * In * 2);
  uint16_t* Wxh_bf = (uint16_t*)carve((size_t)Hn * In * 2);
  uint16_t* Whz_bf = (uint16_t*)carve((size_t)Hn * Hn * 2);
  uint16_t* Whr_bf = (uint16_t*)carve((size_t)Hn * Hn * 2);
  uint16_t* Whh_bf = (uint16_t*)carve((size_t)Hn * Hn * 2);
  float*    xz     = (float*)carve((size_t)Tn * Bn * Hn * 4);
  float*    xr     = (float*)carve((size_t)Tn * Bn * Hn * 4);
  float*    xh     = (float*)carve((size_t)Tn * Bn * Hn * 4);
  float*    hbuf   = (float*)carve((size_t)Bn * Hn * 4);
  float*    zbuf   = (float*)carve((size_t)Bn * Hn * 4);
  uint16_t* h_bf   = (uint16_t*)carve((size_t)Bn * Hn * 2);
  uint16_t* rh_bf  = (uint16_t*)carve((size_t)Bn * Hn * 2);

  // ---- precision conversions (fresh every launch: deterministic) ----
  k_f32_to_bf16<<<2048, 256, 0, stream>>>(x,   x_bf,   Tn * Bn * In);
  k_f32_to_bf16<<<512,  256, 0, stream>>>(Wxz, Wxz_bf, Hn * In);
  k_f32_to_bf16<<<512,  256, 0, stream>>>(Wxr, Wxr_bf, Hn * In);
  k_f32_to_bf16<<<512,  256, 0, stream>>>(Wxh, Wxh_bf, Hn * In);
  k_f32_to_bf16<<<1024, 256, 0, stream>>>(Whz, Whz_bf, Hn * Hn);
  k_f32_to_bf16<<<1024, 256, 0, stream>>>(Whr, Whr_bf, Hn * Hn);
  k_f32_to_bf16<<<1024, 256, 0, stream>>>(Whh, Whh_bf, Hn * Hn);
  k_f32_to_bf16<<<64,   256, 0, stream>>>(h0,  h_bf,   Bn * Hn);
  hipMemcpyAsync(hbuf, h0, (size_t)Bn * Hn * sizeof(float),
                 hipMemcpyDeviceToDevice, stream);

  // ---- phase 1: input projections (parallel WMMA GEMM, 3 gates) ----
  dim3 gx(Tn * Bn / 64, Hn / 128, 3);    // (512, 8, 3)
  k_xproj<<<gx, 256, 0, stream>>>(x_bf, Wxz_bf, Wxr_bf, Wxh_bf,
                                  bxz, bxr, bxh, xz, xr, xh);

  // ---- phase 2: sequential scan, 2 WMMA kernels per step ----
  for (int t = 0; t < Tn; ++t) {
    const float* xz_t = xz + (size_t)t * Bn * Hn;
    const float* xr_t = xr + (size_t)t * Bn * Hn;
    const float* xh_t = xh + (size_t)t * Bn * Hn;
    k_step_zr<<<dim3(Hn / 128, 2), 256, 0, stream>>>(
        h_bf, hbuf, Whz_bf, Whr_bf, bhz, bhr, xz_t, xr_t, zbuf, rh_bf);
    k_step_h<<<dim3(Hn / 128, 1), 256, 0, stream>>>(
        rh_bf, Whh_bf, bhh, xh_t, zbuf, hbuf, h_bf,
        out + (size_t)t * Bn * Hn);
  }

  // h_last == out[T-1]
  hipMemcpyAsync(out + (size_t)Tn * Bn * Hn, out + (size_t)(Tn - 1) * Bn * Hn,
                 (size_t)Bn * Hn * sizeof(float),
                 hipMemcpyDeviceToDevice, stream);
}
